// VAD_Model_77610059039069
// MI455X (gfx1250) — matrix operand order repaired and checked by reading, not compiled, not run on hardware
//
#include <hip/hip_runtime.h>
#include <hip/hip_bf16.h>

// ---------------------------------------------------------------------------
// Mamba-ish VAD model on MI455X (gfx1250), wave32 + WMMA bf16.
// GEMMs (423 GFLOP total) run on v_wmma_f32_16x16x32_bf16 with double-buffered
// async Global->LDS staging (ASYNCcnt) when the toolchain exposes it.
// ---------------------------------------------------------------------------

typedef __attribute__((ext_vector_type(16))) __bf16 bf16x16;
typedef __attribute__((ext_vector_type(8)))  __bf16 bf16x8;
typedef __attribute__((ext_vector_type(8)))  float  v8f;
typedef __attribute__((__vector_size__(4 * sizeof(int)))) int v4i;

#define BM 128
#define BN 128
#define BK 32
#define LDT 40   // padded LDS row stride (bf16): 80B rows -> conflict-free b128 reads

#if __has_builtin(__builtin_amdgcn_global_load_async_to_lds_b128)
#define HAVE_ASYNC_LDS 1
#else
#define HAVE_ASYNC_LDS 0
#endif

#if HAVE_ASYNC_LDS
// builtin signature (from clang diagnostics): (AS1 v4i* src, AS3 v4i* dst, imm offset, imm cpol)
#define ASYNC_CP_B128(gsrc, ldst)                                          \
  __builtin_amdgcn_global_load_async_to_lds_b128(                          \
      (__attribute__((address_space(1))) v4i*)(__bf16*)(gsrc),             \
      (__attribute__((address_space(3))) v4i*)(ldst), 0, 0)
#if __has_builtin(__builtin_amdgcn_s_wait_asynccnt)
#define WAIT_ASYNC_4() __builtin_amdgcn_s_wait_asynccnt(4)
#define WAIT_ASYNC_0() __builtin_amdgcn_s_wait_asynccnt(0)
#else
#define WAIT_ASYNC_4() asm volatile("s_wait_asynccnt 0x4" ::: "memory")
#define WAIT_ASYNC_0() asm volatile("s_wait_asynccnt 0x0" ::: "memory")
#endif
#endif

// ---------------------------------------------------------------------------
// fp32 -> bf16 cast
// ---------------------------------------------------------------------------
__global__ void cast_bf16_kernel(const float* __restrict__ in,
                                 __bf16* __restrict__ out, size_t n) {
  size_t i = (size_t)blockIdx.x * blockDim.x + threadIdx.x;
  if (i < n) out[i] = (__bf16)in[i];
}

// ---------------------------------------------------------------------------
// Tiled bf16 WMMA GEMM:  out[M,N] = A[M,K] @ B[N,K]^T + bias[N]
// Block tile 128x128, 8 waves (2x4), each wave: 4x2 = 8 WMMA 16x16 tiles.
// M % 128 == 0 assumed; N guarded (N=32 / N=128 cases).
// ---------------------------------------------------------------------------
__global__ __launch_bounds__(256) void gemm_bf16_wmma(
    const __bf16* __restrict__ A, const __bf16* __restrict__ Bw,
    const float* __restrict__ bias,
    float* __restrict__ outF, __bf16* __restrict__ outB,
    int M, int N, int K) {
  const int tid  = threadIdx.x;
  const int lane = tid & 31;
  const int wave = tid >> 5;
  const int mBase = blockIdx.y * BM;
  const int nBase = blockIdx.x * BN;
  const int wm = (wave >> 2) * 64;  // 0 / 64
  const int wn = (wave & 3) * 32;   // 0 / 32 / 64 / 96

  // 16-bit A-matrix 16x32 lane layout: lanes 0-15 -> K {0..7,16..23},
  // lanes 16-31 -> K {8..15,24..31}; row = lane & 15.
  const int kHalf = (lane < 16) ? 0 : 8;
  const int fRow  = lane & 15;

  const v8f vzero = {0.f, 0.f, 0.f, 0.f, 0.f, 0.f, 0.f, 0.f};
  v8f acc[4][2];
#pragma unroll
  for (int i = 0; i < 4; ++i)
#pragma unroll
    for (int j = 0; j < 2; ++j) acc[i][j] = vzero;

#if HAVE_ASYNC_LDS
  // -------- async double-buffered staging (ASYNCcnt pipelining) -----------
  __shared__ __bf16 lA[2][BM * LDT];
  __shared__ __bf16 lB[2][BN * LDT];

  // tile = 128 rows x 64B = 512 chunks of 16B; thread covers chunks tid, tid+256
  const int c0 = tid, c1 = tid + 256;
  const int r0 = c0 >> 2, q0 = (c0 & 3) * 8;  // row, bf16 offset within row
  const int r1 = c1 >> 2, q1 = (c1 & 3) * 8;
  const size_t rowA0 = (size_t)(mBase + r0), rowA1 = (size_t)(mBase + r1);
  // clamp B rows past N: garbage lands only in columns >= N (never stored)
  const size_t rowB0 = (size_t)((nBase + r0 < N) ? nBase + r0 : 0);
  const size_t rowB1 = (size_t)((nBase + r1 < N) ? nBase + r1 : 0);

#define ISSUE_STAGE(buf, kk)                                               \
  do {                                                                     \
    ASYNC_CP_B128(A + rowA0 * K + (kk) + q0, &lA[buf][r0 * LDT + q0]);     \
    ASYNC_CP_B128(A + rowA1 * K + (kk) + q1, &lA[buf][r1 * LDT + q1]);     \
    ASYNC_CP_B128(Bw + rowB0 * K + (kk) + q0, &lB[buf][r0 * LDT + q0]);    \
    ASYNC_CP_B128(Bw + rowB1 * K + (kk) + q1, &lB[buf][r1 * LDT + q1]);    \
  } while (0)

  const int nk = K / BK;
  ISSUE_STAGE(0, 0);
  for (int ks = 0; ks < nk; ++ks) {
    const int cur = ks & 1;
    if (ks + 1 < nk) {
      ISSUE_STAGE(cur ^ 1, (ks + 1) * BK);
      WAIT_ASYNC_4();   // stage ks done; stage ks+1 may stay in flight
    } else {
      WAIT_ASYNC_0();
    }
    __syncthreads();

    const __bf16* baseA = &lA[cur][0];
    const __bf16* baseB = &lB[cur][0];
    bf16x16 af[4], bfr[2];
#pragma unroll
    for (int j = 0; j < 2; ++j) {
      const __bf16* p = baseB + (wn + j * 16 + fRow) * LDT + kHalf;
      reinterpret_cast<bf16x8*>(&bfr[j])[0] = *reinterpret_cast<const bf16x8*>(p);
      reinterpret_cast<bf16x8*>(&bfr[j])[1] = *reinterpret_cast<const bf16x8*>(p + 16);
    }
#pragma unroll
    for (int i = 0; i < 4; ++i) {
      const __bf16* p = baseA + (wm + i * 16 + fRow) * LDT + kHalf;
      reinterpret_cast<bf16x8*>(&af[i])[0] = *reinterpret_cast<const bf16x8*>(p);
      reinterpret_cast<bf16x8*>(&af[i])[1] = *reinterpret_cast<const bf16x8*>(p + 16);
    }
#pragma unroll
    for (int i = 0; i < 4; ++i)
#pragma unroll
      for (int j = 0; j < 2; ++j)
        acc[i][j] = __builtin_amdgcn_wmma_f32_16x16x32_bf16(
            false, af[i], false, bfr[j], (short)0, acc[i][j], false, false);
    __syncthreads();
  }
#undef ISSUE_STAGE

#else
  // -------- fallback: synchronous staging, single buffer ------------------
  __shared__ __bf16 lA[BM * LDT];
  __shared__ __bf16 lB[BN * LDT];
  const int sRow  = tid >> 1;
  const int sHalf = (tid & 1) * 16;

  for (int k0 = 0; k0 < K; k0 += BK) {
    {
      const __bf16* src = A + (size_t)(mBase + sRow) * K + k0 + sHalf;
      *reinterpret_cast<uint4*>(&lA[sRow * LDT + sHalf]) =
          *reinterpret_cast<const uint4*>(src);
      *reinterpret_cast<uint4*>(&lA[sRow * LDT + sHalf + 8]) =
          *reinterpret_cast<const uint4*>(src + 8);
    }
    {
      int n = nBase + sRow;
      uint4 v = {0u, 0u, 0u, 0u}, v2 = {0u, 0u, 0u, 0u};
      if (n < N) {
        v = *reinterpret_cast<const uint4*>(Bw + (size_t)n * K + k0 + sHalf);
        v2 = *reinterpret_cast<const uint4*>(Bw + (size_t)n * K + k0 + sHalf + 8);
      }
      *reinterpret_cast<uint4*>(&lB[sRow * LDT + sHalf]) = v;
      *reinterpret_cast<uint4*>(&lB[sRow * LDT + sHalf + 8]) = v2;
    }
    if (k0 + BK < K) {
      __builtin_prefetch(A + (size_t)(mBase + sRow) * K + k0 + BK + sHalf, 0, 1);
      __builtin_prefetch(Bw + (size_t)(nBase + sRow) * K + k0 + BK + sHalf, 0, 1);
    }
    __syncthreads();

    bf16x16 af[4], bfr[2];
#pragma unroll
    for (int j = 0; j < 2; ++j) {
      const __bf16* p = &lB[(wn + j * 16 + fRow) * LDT + kHalf];
      reinterpret_cast<bf16x8*>(&bfr[j])[0] = *reinterpret_cast<const bf16x8*>(p);
      reinterpret_cast<bf16x8*>(&bfr[j])[1] = *reinterpret_cast<const bf16x8*>(p + 16);
    }
#pragma unroll
    for (int i = 0; i < 4; ++i) {
      const __bf16* p = &lA[(wm + i * 16 + fRow) * LDT + kHalf];
      reinterpret_cast<bf16x8*>(&af[i])[0] = *reinterpret_cast<const bf16x8*>(p);
      reinterpret_cast<bf16x8*>(&af[i])[1] = *reinterpret_cast<const bf16x8*>(p + 16);
    }
#pragma unroll
    for (int i = 0; i < 4; ++i)
#pragma unroll
      for (int j = 0; j < 2; ++j)
        acc[i][j] = __builtin_amdgcn_wmma_f32_16x16x32_bf16(
            false, af[i], false, bfr[j], (short)0, acc[i][j], false, false);
    __syncthreads();
  }
#endif

  // epilogue: 16x16 f32 C/D layout (8 VGPRs; lanes 0-15 -> M=r, 16-31 -> M=8+r)
  const int rOff = (lane < 16) ? 0 : 8;
  const int col0 = lane & 15;
#pragma unroll
  for (int i = 0; i < 4; ++i) {
#pragma unroll
    for (int j = 0; j < 2; ++j) {
      int n = nBase + wn + j * 16 + col0;
      if (n >= N) continue;
      float bv = bias ? bias[n] : 0.f;
#pragma unroll
      for (int r = 0; r < 8; ++r) {
        int m = mBase + wm + i * 16 + rOff + r;
        float v = acc[i][j][r] + bv;
        if (outF) outF[(size_t)m * N + n] = v;
        if (outB) outB[(size_t)m * N + n] = (__bf16)v;
      }
    }
  }
}

// ---------------------------------------------------------------------------
// Causal depthwise conv (K=4, left pad 3) + SiLU, reading xh half of xz.
// ---------------------------------------------------------------------------
__global__ void conv_silu_kernel(const __bf16* __restrict__ xz,   // [M, 4096]
                                 const float* __restrict__ w,     // [H*4]
                                 const float* __restrict__ cb,    // [H]
                                 __bf16* __restrict__ out,        // [M, H]
                                 int T, size_t total) {
  size_t idx = (size_t)blockIdx.x * blockDim.x + threadIdx.x;
  if (idx >= total) return;
  const int H = 2048;
  int m = (int)(idx / H), h = (int)(idx % H);
  int t = m % T;
  float acc = cb[h];
#pragma unroll
  for (int k = 0; k < 4; ++k) {
    int dt = k - 3;
    if (t + dt >= 0)
      acc += w[h * 4 + k] * (float)xz[(size_t)(m + dt) * 4096 + h];
  }
  float sig = 1.f / (1.f + __expf(-acc));
  out[idx] = (__bf16)(acc * sig);
}

// ---------------------------------------------------------------------------
// Sequential selective scan: one wave32 per batch, one lane per state dim.
// ---------------------------------------------------------------------------
__global__ __launch_bounds__(32) void scan_kernel(
    const float* __restrict__ xs,    // [B*T*32]
    const float* __restrict__ Amat,  // [32*32]
    const float* __restrict__ Bv, const float* __restrict__ Cv,
    float* __restrict__ sOut,        // [B*T]
    int T) {
  const int b = blockIdx.x;
  const int i = threadIdx.x;
  float Arow[32];
#pragma unroll
  for (int j = 0; j < 32; ++j) Arow[j] = Amat[i * 32 + j];
  float Bi = Bv[i], Ci = Cv[i];
  float h = 0.f;
  for (int t = 0; t < T; ++t) {
    float x = xs[((size_t)b * T + t) * 32 + i];
    float hn = 0.f;
#pragma unroll
    for (int j = 0; j < 32; ++j) hn += Arow[j] * __shfl(h, j, 32);
    h = hn + x * Bi;
    float s = h * Ci;
#pragma unroll
    for (int off = 16; off > 0; off >>= 1) s += __shfl_xor(s, off, 32);
    if (i == 0) sOut[(size_t)b * T + t] = s;
  }
}

// ---------------------------------------------------------------------------
// y = (s[m] + D[h]) * xh_conv * silu(gate);  gate = xz[:, 2048 + h]
// ---------------------------------------------------------------------------
__global__ void gate_kernel(const __bf16* __restrict__ xz,   // [M, 4096]
                            const __bf16* __restrict__ xhc,  // [M, H]
                            const float* __restrict__ s,     // [M]
                            const float* __restrict__ D,     // [H]
                            __bf16* __restrict__ y, size_t total) {
  size_t idx = (size_t)blockIdx.x * blockDim.x + threadIdx.x;
  if (idx >= total) return;
  const int H = 2048;
  int m = (int)(idx / H), h = (int)(idx % H);
  float g = (float)xz[(size_t)m * 4096 + 2048 + h];
  float gs = g / (1.f + __expf(-g));
  float v = (s[m] + D[h]) * (float)xhc[idx] * gs;
  y[idx] = (__bf16)v;
}

// ---------------------------------------------------------------------------
// fc head: logits[m] = <feat[m,:], fc_w> + fc_b   (one wave32 per row)
// ---------------------------------------------------------------------------
__global__ __launch_bounds__(256) void fc_kernel(
    const __bf16* __restrict__ feat, const float* __restrict__ w,
    const float* __restrict__ b, float* __restrict__ out, int Mtot) {
  int lane = threadIdx.x & 31;
  int row = blockIdx.x * (blockDim.x >> 5) + (threadIdx.x >> 5);
  if (row >= Mtot) return;
  float acc = 0.f;
  for (int k = lane; k < 2048; k += 32)
    acc += (float)feat[(size_t)row * 2048 + k] * w[k];
#pragma unroll
  for (int off = 16; off > 0; off >>= 1) acc += __shfl_xor(acc, off, 32);
  if (lane == 0) out[row] = acc + b[0];
}

// ---------------------------------------------------------------------------
// Host launcher
// ---------------------------------------------------------------------------
extern "C" void kernel_launch(void* const* d_in, const int* in_sizes, int n_in,
                              void* d_out, int out_size, void* d_ws, size_t ws_size,
                              hipStream_t stream) {
  const int B = 32, T = 512, Din = 2048, Hid = 2048, DS = 32;
  const int M = B * T;  // 16384

  const float* x         = (const float*)d_in[0];
  const float* in_proj_w = (const float*)d_in[1];
  const float* in_proj_b = (const float*)d_in[2];
  const float* conv_w    = (const float*)d_in[3];
  const float* conv_b    = (const float*)d_in[4];
  const float* ssm_w     = (const float*)d_in[5];
  const float* ssm_b     = (const float*)d_in[6];
  const float* Amat      = (const float*)d_in[7];
  const float* B_ssm     = (const float*)d_in[8];
  const float* C_ssm     = (const float*)d_in[9];
  const float* D_ssm     = (const float*)d_in[10];
  const float* out_w     = (const float*)d_in[11];
  const float* out_b     = (const float*)d_in[12];
  const float* fc_w      = (const float*)d_in[13];
  const float* fc_b      = (const float*)d_in[14];
  const float* proj_w    = (const float*)d_in[15];
  const float* proj_b    = (const float*)d_in[16];
  float* out = (float*)d_out;  // [M] logits then [M,128] z

  char* ws = (char*)d_ws;
  size_t off = 0;
  auto alloc = [&](size_t bytes) {
    size_t r = off;
    off += (bytes + 255) & ~(size_t)255;
    return r;
  };
  __bf16* xb     = (__bf16*)(ws + alloc((size_t)M * Din * 2));
  __bf16* w1b    = (__bf16*)(ws + alloc((size_t)2 * Hid * Din * 2));
  __bf16* xzb    = (__bf16*)(ws + alloc((size_t)M * 2 * Hid * 2));
  __bf16* xhcb   = (__bf16*)(ws + alloc((size_t)M * Hid * 2));
  __bf16* ssmwb  = (__bf16*)(ws + alloc((size_t)DS * Hid * 2));
  float*  xssm   = (float*)(ws + alloc((size_t)M * DS * 4));
  float*  sbuf   = (float*)(ws + alloc((size_t)M * 4));
  __bf16* yb     = (__bf16*)(ws + alloc((size_t)M * Hid * 2));
  __bf16* outwb  = (__bf16*)(ws + alloc((size_t)Din * Hid * 2));
  __bf16* featb  = (__bf16*)(ws + alloc((size_t)M * Din * 2));
  __bf16* projwb = (__bf16*)(ws + alloc((size_t)128 * Din * 2));

  auto cdiv = [](size_t a, size_t b) { return (unsigned)((a + b - 1) / b); };

  // 1. casts to bf16
  cast_bf16_kernel<<<cdiv((size_t)M * Din, 256), 256, 0, stream>>>(x, xb, (size_t)M * Din);
  cast_bf16_kernel<<<cdiv((size_t)2 * Hid * Din, 256), 256, 0, stream>>>(in_proj_w, w1b, (size_t)2 * Hid * Din);
  cast_bf16_kernel<<<cdiv((size_t)DS * Hid, 256), 256, 0, stream>>>(ssm_w, ssmwb, (size_t)DS * Hid);
  cast_bf16_kernel<<<cdiv((size_t)Din * Hid, 256), 256, 0, stream>>>(out_w, outwb, (size_t)Din * Hid);
  cast_bf16_kernel<<<cdiv((size_t)128 * Din, 256), 256, 0, stream>>>(proj_w, projwb, (size_t)128 * Din);

  // 2. in_proj: xz = x @ W1^T + b   [16384 x 4096 x 2048]
  gemm_bf16_wmma<<<dim3((2 * Hid) / BN, M / BM), 256, 0, stream>>>(
      xb, w1b, in_proj_b, nullptr, xzb, M, 2 * Hid, Din);

  // 3. causal depthwise conv + SiLU
  conv_silu_kernel<<<cdiv((size_t)M * Hid, 256), 256, 0, stream>>>(
      xzb, conv_w, conv_b, xhcb, T, (size_t)M * Hid);

  // 4. ssm projection: x_ssm = xh @ ssm_w^T + b   [16384 x 32 x 2048]
  gemm_bf16_wmma<<<dim3(1, M / BM), 256, 0, stream>>>(
      xhcb, ssmwb, ssm_b, xssm, nullptr, M, DS, Hid);

  // 5. sequential selective scan
  scan_kernel<<<B, 32, 0, stream>>>(xssm, Amat, B_ssm, C_ssm, sbuf, T);

  // 6. gate fusion: y = (s + D) * xh_conv * silu(gate)
  gate_kernel<<<cdiv((size_t)M * Hid, 256), 256, 0, stream>>>(
      xzb, xhcb, sbuf, D_ssm, yb, (size_t)M * Hid);

  // 7. out_proj: feat = y @ out_w^T + b   [16384 x 2048 x 2048]
  gemm_bf16_wmma<<<dim3(Din / BN, M / BM), 256, 0, stream>>>(
      yb, outwb, out_b, nullptr, featb, M, Din, Hid);

  // 8. fc head -> logits (first M floats of d_out)
  fc_kernel<<<M / 8, 256, 0, stream>>>(featb, fc_w, fc_b, out, M);

  // 9. proj head: z = feat @ proj_w^T + b   [16384 x 128 x 2048] -> d_out+M
  gemm_bf16_wmma<<<dim3(1, M / BM), 256, 0, stream>>>(
      featb, projwb, proj_b, out + M, nullptr, M, 128, Din);
}
